// MultiHeadAttention_79637283603174
// MI455X (gfx1250) — compile-verified, matrix operand order checked by reading, not run on hardware
//
#include <hip/hip_runtime.h>
#include <hip/hip_bf16.h>

#define HDIM 1024
#define SEQ 2048
#define BATCH 2
#define NHEADS 16
#define HEADD 64
#define MROWS (BATCH * SEQ) // 4096

typedef __bf16 bf16;
typedef __attribute__((ext_vector_type(16))) __bf16 v16bf;
typedef __attribute__((ext_vector_type(8)))  __bf16 v8bf;
typedef __attribute__((ext_vector_type(4)))  __bf16 v4bf;
typedef __attribute__((ext_vector_type(8)))  float  v8f;
typedef __attribute__((ext_vector_type(4)))  unsigned int v4u;
typedef __attribute__((ext_vector_type(8)))  int v8i;
typedef __attribute__((ext_vector_type(4)))  int v4i;

// ---------------------------------------------------------------------------
// WMMA helpers (CDNA5 16x16x32 bf16, f32 accumulate)
// A-fragment (16xK, K-contiguous rows): lane<16 -> row=lane, K halves {0..7,16..23};
// lane>=16 -> row=lane-16, K halves {8..15,24..31}.  B uses same loader with
// "row" = output column (all our B operands are K-contiguous rows).
// ---------------------------------------------------------------------------
__device__ __forceinline__ v16bf load_fragK(const bf16* p0, int stride) {
  const int lane = threadIdx.x & 31;
  const bf16* p = p0 + (size_t)(lane & 15) * stride + ((lane & 16) ? 8 : 0);
  v8bf lo = *(const v8bf*)p;
  v8bf hi = *(const v8bf*)(p + 16);
  v16bf f;
#pragma unroll
  for (int i = 0; i < 8; ++i) { f[i] = lo[i]; f[i + 8] = hi[i]; }
  return f;
}

__device__ __forceinline__ v8f wmma_bf16(v16bf a, v16bf b, v8f c) {
  return __builtin_amdgcn_wmma_f32_16x16x32_bf16(false, a, false, b, (short)0, c,
                                                 false, false);
}

// Escape an LDS pointer so the compiler treats it as written by unknown code
// (the TDM engine). Prevents "never-stored" load folding; zero instructions.
__device__ __forceinline__ void lds_publish(const void* p) {
  asm volatile("" ::"r"(p) : "memory");
}

#if __has_builtin(__builtin_amdgcn_s_wait_tensorcnt)
#define WAIT_TENSORCNT(n) __builtin_amdgcn_s_wait_tensorcnt(n)
#else
#define WAIT_TENSORCNT(n) asm volatile("s_wait_tensorcnt %0" ::"i"(n) : "memory")
#endif

// ---------------------------------------------------------------------------
// TDM: async-load a 2D tile (rows x 32 bf16, row stride HDIM elems) into LDS.
// Builds D# group0/group1 per CDNA5 ISA ch.8 (count=1, type=2 "image",
// data_size=2B, tile_dim0=32, tile_dim1=rows, tensor_dim0_stride=HDIM).
// Tracked by TENSORcnt. Must be issued from one wave (scalar branch).
// 6-arg builtin form: (g0:v4u, g1:v8i, g2:v4i, g3:v4i, extra:v8i, cpol:i32)
// ---------------------------------------------------------------------------
__device__ __forceinline__ void tdm_load_tile(bf16* lds_dst, unsigned int lds_byte_off,
                                              const bf16* gsrc, unsigned int rows) {
#if __has_builtin(__builtin_amdgcn_tensor_load_to_lds)
  const unsigned long long ga = (unsigned long long)(uintptr_t)gsrc;
  v4u g0;
  g0[0] = 1u;                                                     // count=1 (valid)
  g0[1] = lds_byte_off;                                           // lds_addr
  g0[2] = (unsigned int)ga;                                       // global_addr[31:0]
  g0[3] = (unsigned int)((ga >> 32) & 0x01FFFFFFu) | (2u << 30);  // addr[56:32]|type=2
  v8i g1;
  g1[0] = (int)(1u << 16);                 // data_size=1 -> 2 bytes/elem
  g1[1] = (int)(32u << 16);                // tensor_dim0[15:0]=32 (bits 63:48)
  g1[2] = (int)(rows << 16);               // tensor_dim1[15:0]=rows (bits 95:80)
  g1[3] = (int)(32u << 16);                // tile_dim0=32 (bits 127:112)
  g1[4] = (int)rows;                       // tile_dim1=rows, tile_dim2=0
  g1[5] = (int)HDIM;                       // tensor_dim0_stride[31:0]
  g1[6] = 0;                               // stride0[47:32]=0, stride1 lo=0
  g1[7] = 0;
  v4i z4 = {0, 0, 0, 0};
  v8i z8 = {0, 0, 0, 0, 0, 0, 0, 0};
  __builtin_amdgcn_tensor_load_to_lds(g0, g1, z4, z4, z8, 0);
  (void)lds_dst;
#else
  // Fallback: lane-cooperative copy (issuing wave only), 64B per row.
  const int lane = threadIdx.x & 31;
  for (unsigned int r = lane; r < rows; r += 32) {
    v8bf* d = (v8bf*)(lds_dst + (size_t)r * 32);
    const v8bf* s = (const v8bf*)(gsrc + (size_t)r * HDIM);
    d[0] = s[0]; d[1] = s[1]; d[2] = s[2]; d[3] = s[3];
  }
  (void)lds_byte_off;
#endif
}

// ---------------------------------------------------------------------------
// f32 -> bf16 cast (weights)
// ---------------------------------------------------------------------------
__global__ __launch_bounds__(256)
void cast_f32_bf16_kernel(const float* __restrict__ in, bf16* __restrict__ out, int n4) {
  int i = blockIdx.x * 256 + threadIdx.x;
  if (i < n4) {
    const float4 v = ((const float4*)in)[i];
    v4bf o;
    o[0] = (bf16)v.x; o[1] = (bf16)v.y; o[2] = (bf16)v.z; o[3] = (bf16)v.w;
    ((v4bf*)out)[i] = o;
  }
}

// ---------------------------------------------------------------------------
// LayerNorm + bf16 cast (one 1024-wide row per 256-thread block; 8 waves)
// ---------------------------------------------------------------------------
__global__ __launch_bounds__(256)
void ln_cast_kernel(const float* __restrict__ x, const float* __restrict__ gamma,
                    const float* __restrict__ beta, bf16* __restrict__ y) {
  const int row = blockIdx.x;
  const int tid = threadIdx.x;
  const float4 v = ((const float4*)(x + (size_t)row * HDIM))[tid];
  float s  = v.x + v.y + v.z + v.w;
  float s2 = v.x * v.x + v.y * v.y + v.z * v.z + v.w * v.w;
#pragma unroll
  for (int off = 16; off > 0; off >>= 1) {
    s  += __shfl_xor(s, off, 32);
    s2 += __shfl_xor(s2, off, 32);
  }
  __shared__ float sh[2][8];
  if ((tid & 31) == 0) { sh[0][tid >> 5] = s; sh[1][tid >> 5] = s2; }
  __syncthreads();
  float ts = 0.f, ts2 = 0.f;
#pragma unroll
  for (int i = 0; i < 8; ++i) { ts += sh[0][i]; ts2 += sh[1][i]; }
  const float mu  = ts * (1.0f / HDIM);
  const float var = ts2 * (1.0f / HDIM) - mu * mu;
  const float inv = rsqrtf(var + 1e-12f);
  const int c0 = tid * 4;
  const float xv[4] = {v.x, v.y, v.z, v.w};
  v4bf o;
#pragma unroll
  for (int c = 0; c < 4; ++c)
    o[c] = (bf16)((xv[c] - mu) * inv * gamma[c0 + c] + beta[c0 + c]);
  ((v4bf*)(y + (size_t)row * HDIM))[tid] = o;
}

// ---------------------------------------------------------------------------
// TDM-fed, double-buffered WMMA GEMM: C = X[4096x1024] @ W^T + bias (bf16 out).
// Block 256 thr = 8 waves (4 M x 2 N); wave tile 32x32; block tile 128x64.
// Wave 0 streams A(128x32) and B(64x32) tiles into LDS with tensor_load_to_lds,
// overlapped one k-step ahead (TENSORcnt double buffering).
// Static LDS (single object at workgroup LDS offset 0):
//   [0     .. 8192 )  A buf 0   (128x32 bf16)
//   [8192  .. 16384)  A buf 1
//   [16384 .. 20480)  B buf 0   (64x32 bf16)
//   [20480 .. 24576)  B buf 1
// vmode==0 -> [B,NH,S,HD] ; vmode==1 -> [B,NH,HD,S]
// ---------------------------------------------------------------------------
#define A_ELEMS 4096   // 128*32
#define B_ELEMS 2048   // 64*32

__global__ __launch_bounds__(256)
void gemm_qkv_kernel(const bf16* __restrict__ X, const bf16* __restrict__ W,
                     const float* __restrict__ bias, bf16* __restrict__ out,
                     int vmode) {
  __shared__ __align__(16) bf16 smem[2 * A_ELEMS + 2 * B_ELEMS];  // 24576 B
  const int wave = threadIdx.x >> 5;
  const int lane = threadIdx.x & 31;
  const int wid = __builtin_amdgcn_readfirstlane(wave);
  const int wm = wave >> 1, wn = wave & 1;
  const int mb = blockIdx.y * 128;
  const int nb = blockIdx.x * 64;
  const int m0 = mb + wm * 32;
  const int n0 = nb + wn * 32;

  if (wid == 0) {
    tdm_load_tile(smem, 0u, X + (size_t)mb * HDIM, 128);
    tdm_load_tile(smem + 2 * A_ELEMS, 4u * A_ELEMS, W + (size_t)nb * HDIM, 64);
  }

  v8f acc[2][2] = {};
  for (int kk = 0; kk < HDIM / 32; ++kk) {
    const int cur = kk & 1;
    if (wid == 0) {
      if (kk + 1 < HDIM / 32) {
        const int kn = (kk + 1) * 32;
        const int nxt = cur ^ 1;
        tdm_load_tile(smem + nxt * A_ELEMS, (unsigned)(nxt * 2 * A_ELEMS),
                      X + (size_t)mb * HDIM + kn, 128);
        tdm_load_tile(smem + 2 * A_ELEMS + nxt * B_ELEMS,
                      (unsigned)(4 * A_ELEMS + nxt * 2 * B_ELEMS),
                      W + (size_t)nb * HDIM + kn, 64);
        WAIT_TENSORCNT(2);   // current buffer complete; next 2 still in flight
      } else {
        WAIT_TENSORCNT(0);
      }
    }
    __syncthreads();      // cur buffer visible to all waves
    lds_publish(smem);    // TDM wrote LDS: force real ds reads below
    const bf16* At = smem + cur * A_ELEMS;
    const bf16* Bt = smem + 2 * A_ELEMS + cur * B_ELEMS;
    v16bf a0 = load_fragK(At + (wm * 32) * 32, 32);
    v16bf a1 = load_fragK(At + (wm * 32 + 16) * 32, 32);
    v16bf b0 = load_fragK(Bt + (wn * 32) * 32, 32);
    v16bf b1 = load_fragK(Bt + (wn * 32 + 16) * 32, 32);
    acc[0][0] = wmma_bf16(a0, b0, acc[0][0]);
    acc[0][1] = wmma_bf16(a0, b1, acc[0][1]);
    acc[1][0] = wmma_bf16(a1, b0, acc[1][0]);
    acc[1][1] = wmma_bf16(a1, b1, acc[1][1]);
    __syncthreads();  // all reads of cur done before it is overwritten
  }

  const int col = lane & 15;
  const int r0 = (lane & 16) ? 8 : 0;
#pragma unroll
  for (int i = 0; i < 2; ++i)
#pragma unroll
    for (int jj = 0; jj < 2; ++jj)
#pragma unroll
      for (int e = 0; e < 8; ++e) {
        const int m = m0 + i * 16 + r0 + e;
        const int n = n0 + jj * 16 + col;
        const float val = acc[i][jj][e] + bias[n];
        const int bb = m >> 11;          // m / SEQ
        const int s  = m & (SEQ - 1);
        const int hh = n >> 6;           // n / HEADD
        const int d  = n & (HEADD - 1);
        size_t idx;
        if (vmode)
          idx = (((size_t)(bb * NHEADS + hh)) * HEADD + d) * SEQ + s;
        else
          idx = (((size_t)(bb * NHEADS + hh)) * SEQ + s) * HEADD + d;
        out[idx] = (bf16)val;
      }
}

// ---------------------------------------------------------------------------
// Output GEMM (same TDM pipeline): out = CTX @ Wo^T + bo + residual (f32 store)
// ---------------------------------------------------------------------------
__global__ __launch_bounds__(256)
void gemm_out_kernel(const bf16* __restrict__ X, const bf16* __restrict__ W,
                     const float* __restrict__ bias, const float* __restrict__ residual,
                     float* __restrict__ out) {
  __shared__ __align__(16) bf16 smem[2 * A_ELEMS + 2 * B_ELEMS];  // 24576 B
  const int wave = threadIdx.x >> 5;
  const int lane = threadIdx.x & 31;
  const int wid = __builtin_amdgcn_readfirstlane(wave);
  const int wm = wave >> 1, wn = wave & 1;
  const int mb = blockIdx.y * 128;
  const int nb = blockIdx.x * 64;
  const int m0 = mb + wm * 32;
  const int n0 = nb + wn * 32;

  if (wid == 0) {
    tdm_load_tile(smem, 0u, X + (size_t)mb * HDIM, 128);
    tdm_load_tile(smem + 2 * A_ELEMS, 4u * A_ELEMS, W + (size_t)nb * HDIM, 64);
  }

  v8f acc[2][2] = {};
  for (int kk = 0; kk < HDIM / 32; ++kk) {
    const int cur = kk & 1;
    if (wid == 0) {
      if (kk + 1 < HDIM / 32) {
        const int kn = (kk + 1) * 32;
        const int nxt = cur ^ 1;
        tdm_load_tile(smem + nxt * A_ELEMS, (unsigned)(nxt * 2 * A_ELEMS),
                      X + (size_t)mb * HDIM + kn, 128);
        tdm_load_tile(smem + 2 * A_ELEMS + nxt * B_ELEMS,
                      (unsigned)(4 * A_ELEMS + nxt * 2 * B_ELEMS),
                      W + (size_t)nb * HDIM + kn, 64);
        WAIT_TENSORCNT(2);
      } else {
        WAIT_TENSORCNT(0);
      }
    }
    __syncthreads();
    lds_publish(smem);
    const bf16* At = smem + cur * A_ELEMS;
    const bf16* Bt = smem + 2 * A_ELEMS + cur * B_ELEMS;
    v16bf a0 = load_fragK(At + (wm * 32) * 32, 32);
    v16bf a1 = load_fragK(At + (wm * 32 + 16) * 32, 32);
    v16bf b0 = load_fragK(Bt + (wn * 32) * 32, 32);
    v16bf b1 = load_fragK(Bt + (wn * 32 + 16) * 32, 32);
    acc[0][0] = wmma_bf16(a0, b0, acc[0][0]);
    acc[0][1] = wmma_bf16(a0, b1, acc[0][1]);
    acc[1][0] = wmma_bf16(a1, b0, acc[1][0]);
    acc[1][1] = wmma_bf16(a1, b1, acc[1][1]);
    __syncthreads();
  }

  const int col = lane & 15;
  const int r0 = (lane & 16) ? 8 : 0;
#pragma unroll
  for (int i = 0; i < 2; ++i)
#pragma unroll
    for (int jj = 0; jj < 2; ++jj)
#pragma unroll
      for (int e = 0; e < 8; ++e) {
        const int m = m0 + i * 16 + r0 + e;
        const int n = n0 + jj * 16 + col;
        const size_t idx = (size_t)m * HDIM + n;
        out[idx] = acc[i][jj][e] + bias[n] + residual[idx];
      }
}

// ---------------------------------------------------------------------------
// Flash attention: block = 4 waves, each wave owns a 16-query tile of one
// (b,h). Streams 32 keys per step: 4 WMMAs for scores, online softmax with
// 16-lane shfl reductions, P staged through per-wave LDS tile (C-layout ->
// A-layout), 4 WMMAs for P @ V^T.  ctx written bf16 as [B,S,H].
// ---------------------------------------------------------------------------
__global__ __launch_bounds__(128)
void attn_kernel(const bf16* __restrict__ Q, const bf16* __restrict__ K,
                 const bf16* __restrict__ VT, const float* __restrict__ mask,
                 bf16* __restrict__ ctx) {
  const int bh = blockIdx.x;
  const int b = bh >> 4;
  const int h = bh & 15;
  const int wave = threadIdx.x >> 5;
  const int lane = threadIdx.x & 31;
  const int q0 = blockIdx.y * 64 + wave * 16;

  const bf16* Qb = Q + ((size_t)bh * SEQ + q0) * HEADD;
  const bf16* Kb = K + (size_t)bh * SEQ * HEADD;
  const bf16* Vb = VT + (size_t)bh * HEADD * SEQ;
  const float* mb = mask + (size_t)b * SEQ;

  const v16bf qf0 = load_fragK(Qb, HEADD);
  const v16bf qf1 = load_fragK(Qb + 32, HEADD);

  float mrow[8], lrow[8];
#pragma unroll
  for (int e = 0; e < 8; ++e) { mrow[e] = -3.0e38f; lrow[e] = 0.f; }
  v8f acc[4] = {};

  __shared__ __align__(16) bf16 pbuf[4][16 * 32];
  bf16* pb = pbuf[wave];
  const int col = lane & 15;
  const int r0 = (lane & 16) ? 8 : 0;

  for (int j = 0; j < SEQ; j += 32) {
    v16bf kf00 = load_fragK(Kb + (size_t)j * HEADD, HEADD);
    v16bf kf01 = load_fragK(Kb + (size_t)j * HEADD + 32, HEADD);
    v16bf kf10 = load_fragK(Kb + (size_t)(j + 16) * HEADD, HEADD);
    v16bf kf11 = load_fragK(Kb + (size_t)(j + 16) * HEADD + 32, HEADD);
    v8f s0 = {}, s1 = {};
    s0 = wmma_bf16(qf0, kf00, s0);
    s0 = wmma_bf16(qf1, kf01, s0);
    s1 = wmma_bf16(qf0, kf10, s1);
    s1 = wmma_bf16(qf1, kf11, s1);

    const float mk0 = mb[j + col];
    const float mk1 = mb[j + 16 + col];
    float p0[8], p1[8], rm[8];
#pragma unroll
    for (int e = 0; e < 8; ++e) {
      p0[e] = s0[e] * 0.125f + mk0;   // 1/sqrt(64)
      p1[e] = s1[e] * 0.125f + mk1;
      rm[e] = fmaxf(p0[e], p1[e]);
    }
#pragma unroll
    for (int off = 1; off < 16; off <<= 1)
#pragma unroll
      for (int e = 0; e < 8; ++e)
        rm[e] = fmaxf(rm[e], __shfl_xor(rm[e], off, 32));

    float alpha[8], rs[8];
#pragma unroll
    for (int e = 0; e < 8; ++e) {
      const float mnew = fmaxf(mrow[e], rm[e]);
      alpha[e] = __expf(mrow[e] - mnew);
      mrow[e] = mnew;
      p0[e] = __expf(p0[e] - mnew);
      p1[e] = __expf(p1[e] - mnew);
      rs[e] = p0[e] + p1[e];
    }
#pragma unroll
    for (int off = 1; off < 16; off <<= 1)
#pragma unroll
      for (int e = 0; e < 8; ++e)
        rs[e] += __shfl_xor(rs[e], off, 32);
#pragma unroll
    for (int e = 0; e < 8; ++e) {
      lrow[e] = lrow[e] * alpha[e] + rs[e];
#pragma unroll
      for (int g = 0; g < 4; ++g) acc[g][e] *= alpha[e];
    }

#pragma unroll
    for (int e = 0; e < 8; ++e) {
      pb[(r0 + e) * 32 + col]      = (bf16)p0[e];
      pb[(r0 + e) * 32 + col + 16] = (bf16)p1[e];
    }
    asm volatile("s_wait_dscnt 0" ::: "memory");  // wave-synchronous LDS round-trip
    v16bf pf = load_fragK(pb, 32);

#pragma unroll
    for (int g = 0; g < 4; ++g) {
      v16bf vf = load_fragK(Vb + (size_t)(g * 16) * SEQ + j, SEQ);
      acc[g] = wmma_bf16(pf, vf, acc[g]);
    }
  }

#pragma unroll
  for (int g = 0; g < 4; ++g)
#pragma unroll
    for (int e = 0; e < 8; ++e) {
      const float o = acc[g][e] / lrow[e];
      const int srow = q0 + r0 + e;
      ctx[((size_t)(b * SEQ + srow)) * HDIM + h * HEADD + g * 16 + col] = (bf16)o;
    }
}

// ---------------------------------------------------------------------------
extern "C" void kernel_launch(void* const* d_in, const int* in_sizes, int n_in,
                              void* d_out, int out_size, void* d_ws, size_t ws_size,
                              hipStream_t stream) {
  (void)in_sizes; (void)n_in; (void)out_size; (void)ws_size;
  const float* hs   = (const float*)d_in[0];
  const float* mask = (const float*)d_in[1];
  const float* wq = (const float*)d_in[2];
  const float* bq = (const float*)d_in[3];
  const float* wk = (const float*)d_in[4];
  const float* bk = (const float*)d_in[5];
  const float* wv = (const float*)d_in[6];
  const float* bv = (const float*)d_in[7];
  const float* wo = (const float*)d_in[8];
  const float* bo = (const float*)d_in[9];
  const float* lng = (const float*)d_in[10];
  const float* lnb = (const float*)d_in[11];
  float* out = (float*)d_out;

  char* ws = (char*)d_ws;
  const size_t MB = 1024 * 1024;
  bf16* xln = (bf16*)(ws);             // 8 MB  x_ln bf16 [4096,1024]
  bf16* wqb = (bf16*)(ws + 8 * MB);    // 2 MB each
  bf16* wkb = (bf16*)(ws + 10 * MB);
  bf16* wvb = (bf16*)(ws + 12 * MB);
  bf16* wob = (bf16*)(ws + 14 * MB);
  bf16* Qb  = (bf16*)(ws + 16 * MB);   // 8 MB  [B,NH,S,HD]
  bf16* Kb  = (bf16*)(ws + 24 * MB);   // 8 MB  [B,NH,S,HD]
  bf16* VTb = (bf16*)(ws + 32 * MB);   // 8 MB  [B,NH,HD,S]
  bf16* ctx = (bf16*)(ws + 40 * MB);   // 8 MB  [B,S,H]

  const int n4w = (HDIM * HDIM) / 4;   // 262144 float4 groups per weight
  cast_f32_bf16_kernel<<<n4w / 256, 256, 0, stream>>>(wq, wqb, n4w);
  cast_f32_bf16_kernel<<<n4w / 256, 256, 0, stream>>>(wk, wkb, n4w);
  cast_f32_bf16_kernel<<<n4w / 256, 256, 0, stream>>>(wv, wvb, n4w);
  cast_f32_bf16_kernel<<<n4w / 256, 256, 0, stream>>>(wo, wob, n4w);

  ln_cast_kernel<<<MROWS, 256, 0, stream>>>(hs, lng, lnb, xln);

  dim3 ggrid(HDIM / 64, MROWS / 128);  // (16, 32)
  gemm_qkv_kernel<<<ggrid, 256, 0, stream>>>(xln, wqb, bq, Qb, 0);
  gemm_qkv_kernel<<<ggrid, 256, 0, stream>>>(xln, wkb, bk, Kb, 0);
  gemm_qkv_kernel<<<ggrid, 256, 0, stream>>>(xln, wvb, bv, VTb, 1);

  dim3 agrid(BATCH * NHEADS, SEQ / 64); // (32, 32)
  attn_kernel<<<agrid, 128, 0, stream>>>(Qb, Kb, VTb, mask, ctx);

  gemm_out_kernel<<<ggrid, 256, 0, stream>>>(ctx, wob, bo, hs, out);
}
